// ResidualAttentionBlock_27968827032246
// MI455X (gfx1250) — matrix operand order repaired and checked by reading, compiled, not run on hardware
//
#include <hip/hip_runtime.h>
#include <hip/hip_bf16.h>
#include <math.h>

// Workspace requirement: ~540 MB (aliased buffers; MI455X has 432 GiB HBM).

#define NH_ 12
#define USE_ASYNC_COPY 1

typedef __attribute__((ext_vector_type(16))) _Float16 v16h;
typedef __attribute__((ext_vector_type(8)))  float    v8f;

union HF16 { v16h v; _Float16 h[16]; unsigned int u[8]; };

__device__ __forceinline__ float gelu_exact(float x) {
  return 0.5f * x * (1.0f + erff(x * 0.70710678118654752f));
}
__device__ __forceinline__ float silu1702(float x) {
  return x / (1.0f + __expf(-1.702f * x));
}

__device__ __forceinline__ float epi(float v, int ACT, float bv, float scale) {
  v += bv;
  if (ACT == 1) v = gelu_exact(v);
  else if (ACT == 2) v = silu1702(v);
  return v * scale;
}

// ---------------------------------------------------------------------------
// Large WMMA GEMM: out[M,N] = epilogue( A[M,K](f16) @ W[N,K]^T(f16) )
// block = 256 threads (8 waves), tile 128(M) x 128(N) x 32(K).
// Wave (wm,wn) in 4x2 grid computes 32x64 via 2 A-frags x 4 B-frags = 8 WMMA.
// Requires M%128==0, N%128==0, K%32==0.
// Tile loads: GLOBAL_LOAD_ASYNC_TO_LDS_B128 (ASYNCcnt), 16B-aligned.
// ---------------------------------------------------------------------------
template <int ACT>
__global__ void __launch_bounds__(256)
gemm128(const _Float16* __restrict__ A, const _Float16* __restrict__ W,
        const float* __restrict__ bias,
        const float* __restrict__ res1, const float* __restrict__ res2,
        float* __restrict__ outF, _Float16* __restrict__ outH,
        int M, int N, int K, float scale) {
  __shared__ __align__(16) _Float16 As[128 * 40];
  __shared__ __align__(16) _Float16 Bs[128 * 40];
  const int tid  = threadIdx.x;
  const int wave = tid >> 5;
  const int lane = tid & 31;
  const int hi   = (lane >> 4) & 1;
  const int ln16 = lane & 15;
  const int wm   = wave & 3;        // 4 row-strips of 32
  const int wn   = wave >> 2;       // 2 col-strips of 64
  const int m0   = blockIdx.y * 128;
  const int n0   = blockIdx.x * 128;

  v8f acc[8] = {};

  const unsigned int* Ag = (const unsigned int*)A;
  const unsigned int* Wg = (const unsigned int*)W;
  unsigned int* AsU = (unsigned int*)As;
  unsigned int* BsU = (unsigned int*)Bs;
  const int K2 = K >> 1;
#if USE_ASYNC_COPY
  const unsigned asBase = (unsigned)(unsigned long long)(void*)AsU;
  const unsigned bsBase = (unsigned)(unsigned long long)(void*)BsU;
#endif

  for (int k0 = 0; k0 < K; k0 += 32) {
    __syncthreads();
#if USE_ASYNC_COPY
    // 512 b128 chunks per tile; 2 per thread per tile.
#pragma unroll
    for (int i = 0; i < 2; i++) {
      int idx = tid + i * 256;            // 0..511
      int r = idx >> 2, q = idx & 3;      // row, 16B quarter
      unsigned long long ga =
          (unsigned long long)(const void*)(Ag + (size_t)(m0 + r) * K2 + (k0 >> 1) + q * 4);
      unsigned long long gb =
          (unsigned long long)(const void*)(Wg + (size_t)(n0 + r) * K2 + (k0 >> 1) + q * 4);
      unsigned la = asBase + (unsigned)(r * 80 + q * 16);
      unsigned lb = bsBase + (unsigned)(r * 80 + q * 16);
      asm volatile("global_load_async_to_lds_b128 %0, %1, off"
                   :: "v"(la), "v"(ga) : "memory");
      asm volatile("global_load_async_to_lds_b128 %0, %1, off"
                   :: "v"(lb), "v"(gb) : "memory");
    }
    asm volatile("s_wait_asynccnt 0x0" ::: "memory");
#else
#pragma unroll
    for (int i = 0; i < 8; i++) {
      int u = tid + i * 256;              // 0..2047
      int r = u >> 4, c = u & 15;         // 128 rows x 16 uints
      AsU[r * 20 + c] = Ag[(size_t)(m0 + r) * K2 + (k0 >> 1) + c];
      BsU[r * 20 + c] = Wg[(size_t)(n0 + r) * K2 + (k0 >> 1) + c];
    }
#endif
    __syncthreads();

    // A fragments (16x32 f16, gfx1250 layout)
    HF16 a[2];
#pragma unroll
    for (int s = 0; s < 2; s++) {
      const int arow = wm * 32 + s * 16 + ln16;
#pragma unroll
      for (int p = 0; p < 8; p++) {
        int k = (p < 4 ? 2 * p : 2 * p + 8) + (hi ? 8 : 0);
        a[s].u[p] = AsU[arow * 20 + (k >> 1)];
      }
    }
#pragma unroll
    for (int nt = 0; nt < 4; nt++) {
      HF16 b;
      const int bcol = wn * 64 + nt * 16 + ln16;
#pragma unroll
      for (int p = 0; p < 8; p++) {
        int k = (p < 4 ? 2 * p : 2 * p + 8) + (hi ? 8 : 0);
        b.u[p] = BsU[bcol * 20 + (k >> 1)];
      }
#pragma unroll
      for (int s = 0; s < 2; s++)
        acc[s * 4 + nt] = __builtin_amdgcn_wmma_f32_16x16x32_f16(
            false, a[s].v, false, b.v, (short)0, acc[s * 4 + nt], false, false);
    }
  }

#pragma unroll
  for (int s = 0; s < 2; s++) {
#pragma unroll
    for (int nt = 0; nt < 4; nt++) {
      const int col = n0 + wn * 64 + nt * 16 + ln16;
      const float bv = bias ? bias[col] : 0.0f;
#pragma unroll
      for (int r = 0; r < 8; r++) {
        const int row = m0 + wm * 32 + s * 16 + r + (hi ? 8 : 0);
        float v = epi(acc[s * 4 + nt][r], ACT, bv, scale);
        const size_t off = (size_t)row * N + col;
        if (res1) v += res1[off];
        if (res2) v += res2[off];
        if (outF) outF[off] = v;
        if (outH) outH[off] = (_Float16)v;
      }
    }
  }
}

// ---------------------------------------------------------------------------
// Small WMMA GEMM (N=192 adapters): block 128 threads, tile 64x64x32.
// ---------------------------------------------------------------------------
template <int ACT>
__global__ void __launch_bounds__(128)
gemm64(const _Float16* __restrict__ A, const _Float16* __restrict__ W,
       const float* __restrict__ bias,
       float* __restrict__ outF, _Float16* __restrict__ outH,
       int M, int N, int K, float scale) {
  __shared__ __align__(16) _Float16 As[64 * 40];
  __shared__ __align__(16) _Float16 Bs[64 * 40];
  const int tid  = threadIdx.x;
  const int wave = tid >> 5;
  const int lane = tid & 31;
  const int hi   = (lane >> 4) & 1;
  const int ln16 = lane & 15;
  const int m0   = blockIdx.y * 64;
  const int n0   = blockIdx.x * 64;

  v8f acc[4] = {};
  const unsigned int* Ag = (const unsigned int*)A;
  const unsigned int* Wg = (const unsigned int*)W;
  unsigned int* AsU = (unsigned int*)As;
  unsigned int* BsU = (unsigned int*)Bs;
  const int K2 = K >> 1;

  for (int k0 = 0; k0 < K; k0 += 32) {
    __syncthreads();
#pragma unroll
    for (int i = 0; i < 8; i++) {
      int u = tid + i * 128;
      int r = u >> 4, c = u & 15;
      AsU[r * 20 + c] = Ag[(size_t)(m0 + r) * K2 + (k0 >> 1) + c];
      BsU[r * 20 + c] = Wg[(size_t)(n0 + r) * K2 + (k0 >> 1) + c];
    }
    __syncthreads();

    HF16 a;
    const int arow = wave * 16 + ln16;
#pragma unroll
    for (int p = 0; p < 8; p++) {
      int k = (p < 4 ? 2 * p : 2 * p + 8) + (hi ? 8 : 0);
      a.u[p] = AsU[arow * 20 + (k >> 1)];
    }
#pragma unroll
    for (int nt = 0; nt < 4; nt++) {
      HF16 b;
      const int bcol = nt * 16 + ln16;
#pragma unroll
      for (int p = 0; p < 8; p++) {
        int k = (p < 4 ? 2 * p : 2 * p + 8) + (hi ? 8 : 0);
        b.u[p] = BsU[bcol * 20 + (k >> 1)];
      }
      acc[nt] = __builtin_amdgcn_wmma_f32_16x16x32_f16(
          false, a.v, false, b.v, (short)0, acc[nt], false, false);
    }
  }

#pragma unroll
  for (int nt = 0; nt < 4; nt++) {
    const int col = n0 + nt * 16 + ln16;
    const float bv = bias ? bias[col] : 0.0f;
#pragma unroll
    for (int r = 0; r < 8; r++) {
      const int row = m0 + wave * 16 + r + (hi ? 8 : 0);
      float v = epi(acc[nt][r], ACT, bv, scale);
      const size_t off = (size_t)row * N + col;
      if (outF) outF[off] = v;
      if (outH) outH[off] = (_Float16)v;
    }
  }
}

// ---------------------------------------------------------------------------
// Flash attention: per (n,h), Tq queries over Tk keys, HD=64.
// Token row for time t is (t*Nb + n). grid.x = Nb*NH, grid.y = ceil(Tq/64).
// block = 128 threads (4 waves), each wave owns 16 query rows.
// ---------------------------------------------------------------------------
__global__ void __launch_bounds__(128)
flash_attn(const _Float16* __restrict__ Q, int qStride,
           const _Float16* __restrict__ Kp, int kStride,
           const _Float16* __restrict__ Vp, int vStride,
           _Float16* __restrict__ O, int oStride,
           int Nb, int Tq, int Tk, float scale) {
  __shared__ __align__(16) _Float16 Qs[64 * 72];
  __shared__ __align__(16) _Float16 Ks[32 * 72];
  __shared__ __align__(16) _Float16 Vs[32 * 72];
  __shared__ __align__(16) _Float16 Ps[4 * 16 * 36];

  const int tid  = threadIdx.x;
  const int wave = tid >> 5;
  const int lane = tid & 31;
  const int hi   = (lane >> 4) & 1;
  const int ln   = lane & 15;
  const int n    = blockIdx.x % Nb;
  const int h    = blockIdx.x / Nb;
  const int qbase = blockIdx.y * 64;

  unsigned int* QsU = (unsigned int*)Qs;
  unsigned int* KsU = (unsigned int*)Ks;
  unsigned int* VsU = (unsigned int*)Vs;

#pragma unroll
  for (int i = 0; i < 16; i++) {
    int u = tid + i * 128;
    int r = u >> 5, c = u & 31;
    int t = qbase + r;
    unsigned int val = 0u;
    if (t < Tq)
      val = *(const unsigned int*)(Q + (size_t)(t * Nb + n) * qStride + h * 64 + 2 * c);
    QsU[r * 36 + c] = val;
  }
  __syncthreads();

  HF16 a0, a1;
  {
    const int qr = wave * 16 + ln;
#pragma unroll
    for (int p = 0; p < 8; p++) {
      int k = (p < 4 ? 2 * p : 2 * p + 8) + (hi ? 8 : 0);
      a0.u[p] = QsU[qr * 36 + (k >> 1)];
      a1.u[p] = QsU[qr * 36 + ((k + 32) >> 1)];
    }
  }

  float mrow[8], lrow[8];
#pragma unroll
  for (int r = 0; r < 8; r++) { mrow[r] = -1e30f; lrow[r] = 0.0f; }
  v8f o[4] = {};

  const int nsteps = (Tk + 31) >> 5;
  for (int kb = 0; kb < nsteps; kb++) {
    const int kbase = kb * 32;
    __syncthreads();
#pragma unroll
    for (int i = 0; i < 8; i++) {
      int u = tid + i * 128;
      int r = u >> 5, c = u & 31;
      int tk = kbase + r;
      unsigned int kv = 0u, vv = 0u;
      if (tk < Tk) {
        kv = *(const unsigned int*)(Kp + (size_t)(tk * Nb + n) * kStride + h * 64 + 2 * c);
        vv = *(const unsigned int*)(Vp + (size_t)(tk * Nb + n) * vStride + h * 64 + 2 * c);
      }
      KsU[r * 36 + c] = kv;
      VsU[r * 36 + c] = vv;
    }
    __syncthreads();

    v8f sfr[2];
#pragma unroll
    for (int nt = 0; nt < 2; nt++) {
      v8f sa = {};
#pragma unroll
      for (int cs = 0; cs < 2; cs++) {
        HF16 bk;
        const int key = nt * 16 + ln;
#pragma unroll
        for (int p = 0; p < 8; p++) {
          int dl = 32 * cs + (p < 4 ? 2 * p : 2 * p + 8) + (hi ? 8 : 0);
          bk.u[p] = KsU[key * 36 + (dl >> 1)];
        }
        sa = __builtin_amdgcn_wmma_f32_16x16x32_f16(
            false, (cs ? a1.v : a0.v), false, bk.v, (short)0, sa, false, false);
      }
      sfr[nt] = sa;
    }

    const int key0 = kbase + ln;
    const int key1 = kbase + 16 + ln;
#pragma unroll
    for (int r = 0; r < 8; r++) {
      sfr[0][r] = (key0 < Tk) ? sfr[0][r] * scale : -1e30f;
      sfr[1][r] = (key1 < Tk) ? sfr[1][r] * scale : -1e30f;
    }

#pragma unroll
    for (int r = 0; r < 8; r++) {
      float rm = fmaxf(sfr[0][r], sfr[1][r]);
#pragma unroll
      for (int off = 8; off >= 1; off >>= 1) rm = fmaxf(rm, __shfl_xor(rm, off, 32));
      float mnew  = fmaxf(mrow[r], rm);
      float alpha = __expf(mrow[r] - mnew);
      float p0 = __expf(sfr[0][r] - mnew);
      float p1 = __expf(sfr[1][r] - mnew);
      float rs = p0 + p1;
#pragma unroll
      for (int off = 8; off >= 1; off >>= 1) rs += __shfl_xor(rs, off, 32);
      lrow[r] = lrow[r] * alpha + rs;
      mrow[r] = mnew;
#pragma unroll
      for (int nt = 0; nt < 4; nt++) o[nt][r] *= alpha;
      const int prow = r + (hi ? 8 : 0);
      Ps[wave * 576 + prow * 36 + ln]      = (_Float16)p0;
      Ps[wave * 576 + prow * 36 + 16 + ln] = (_Float16)p1;
    }
    __syncthreads();

    HF16 pa;
#pragma unroll
    for (int p = 0; p < 8; p++) {
      int k = (p < 4 ? 2 * p : 2 * p + 8) + (hi ? 8 : 0);
      pa.u[p] = *(const unsigned int*)&Ps[wave * 576 + ln * 36 + k];
    }
#pragma unroll
    for (int nt = 0; nt < 4; nt++) {
      HF16 bv;
      const int dcol = nt * 16 + ln;
#pragma unroll
      for (int j = 0; j < 16; j++) {
        int kk = (j < 8 ? j : j + 8) + (hi ? 8 : 0);
        bv.h[j] = Vs[kk * 72 + dcol];
      }
      o[nt] = __builtin_amdgcn_wmma_f32_16x16x32_f16(
          false, pa.v, false, bv.v, (short)0, o[nt], false, false);
    }
  }

#pragma unroll
  for (int r = 0; r < 8; r++) {
    const int t = qbase + wave * 16 + r + (hi ? 8 : 0);
    if (t < Tq) {
      const float inv = lrow[r] > 0.0f ? 1.0f / lrow[r] : 0.0f;
#pragma unroll
      for (int nt = 0; nt < 4; nt++)
        O[(size_t)(t * Nb + n) * oStride + h * 64 + nt * 16 + ln] =
            (_Float16)(o[nt][r] * inv);
    }
  }
}

// ---------------------------------------------------------------------------
// LayerNorm kernels: one wave32 per 768-vector (24 elems/lane).
// ---------------------------------------------------------------------------
__device__ __forceinline__ void ln_core(const float* src, const float* g,
                                        const float* b, _Float16* dst, int lane) {
  float vals[24], s = 0.0f, s2 = 0.0f;
#pragma unroll
  for (int i = 0; i < 24; i++) {
    float v = src[lane + 32 * i];
    vals[i] = v; s += v; s2 += v * v;
  }
#pragma unroll
  for (int off = 16; off >= 1; off >>= 1) {
    s  += __shfl_xor(s,  off, 32);
    s2 += __shfl_xor(s2, off, 32);
  }
  const float mean = s * (1.0f / 768.0f);
  const float var  = s2 * (1.0f / 768.0f) - mean * mean;
  const float rstd = rsqrtf(var + 1e-5f);
#pragma unroll
  for (int i = 0; i < 24; i++) {
    int d = lane + 32 * i;
    dst[d] = (_Float16)((vals[i] - mean) * rstd * g[d] + b[d]);
  }
}

__global__ void __launch_bounds__(256)
ln_temporal_k(const float* __restrict__ x, const float* __restrict__ g,
              const float* __restrict__ b, _Float16* __restrict__ out) {
  const int tok = blockIdx.x * 8 + (threadIdx.x >> 5);  // 0..127  (t*16+bb)
  if (tok >= 128) return;
  const int t = tok / 16, bb = tok % 16;
  ln_core(x + (size_t)(bb * 8 + t) * 768, g, b, out + (size_t)tok * 768,
          threadIdx.x & 31);
}

__global__ void __launch_bounds__(256)
ln1_k(const float* __restrict__ xbuf, const float* __restrict__ g,
      const float* __restrict__ b, _Float16* __restrict__ out, int ntok) {
  const int tok = blockIdx.x * 8 + (threadIdx.x >> 5);
  if (tok >= ntok) return;
  ln_core(xbuf + (size_t)tok * 768, g, b, out + (size_t)tok * 768, threadIdx.x & 31);
}

__global__ void __launch_bounds__(256)
ln2_copy_k(const float* __restrict__ xbuf, const float* __restrict__ g,
           const float* __restrict__ b, float* __restrict__ outF,
           _Float16* __restrict__ outH, int ntok) {
  const int tok = blockIdx.x * 8 + (threadIdx.x >> 5);
  if (tok >= ntok) return;
  const int lane = threadIdx.x & 31;
  const int row2 = tok / 128, c = tok % 128;
  const int srow = (row2 == 0) ? 0 : row2 + 1;
  const float* src = xbuf + (size_t)(srow * 128 + c) * 768;
  float* dcp = outF + (size_t)tok * 768;
#pragma unroll
  for (int i = 0; i < 24; i++) dcp[lane + 32 * i] = src[lane + 32 * i];
  ln_core(src, g, b, outH + (size_t)tok * 768, lane);
}

// ---------------------------------------------------------------------------
// Auxiliary elementwise kernels
// ---------------------------------------------------------------------------
__global__ void __launch_bounds__(256)
conv_f16_k(const float* __restrict__ src, _Float16* __restrict__ dst, long n) {
  long i = (long)blockIdx.x * 256 + threadIdx.x;
  if (i < n) dst[i] = (_Float16)src[i];
}

__global__ void __launch_bounds__(256)
build_xbuf_k(const float* __restrict__ x, const float* __restrict__ xt2,
             float* __restrict__ xbuf) {
  long i = (long)blockIdx.x * 256 + threadIdx.x;
  const long n = (long)198 * 128 * 768;
  if (i >= n) return;
  const int row = (int)(i / (128 * 768));
  const int rem = (int)(i % (128 * 768));
  float v;
  if (row == 0) v = x[rem];
  else if (row == 1) {
    const int c = rem / 768, d = rem % 768;
    v = xt2[(size_t)((c % 8) * 16 + c / 8) * 768 + d];
  } else v = x[(size_t)(row - 1) * 128 * 768 + rem];
  xbuf[i] = v;
}

__global__ void __launch_bounds__(256)
shift_k(_Float16* __restrict__ kv) {
  long i = (long)blockIdx.x * 256 + threadIdx.x;
  const long n = (long)196 * 128 * 768;
  if (i >= n) return;
  const int pr  = (int)(i / (128 * 768));
  const int rem = (int)(i % (128 * 768));
  const int c = rem / 768, d = rem % 768;
  const int bb = c / 8, t = c & 7;
  const int ci = (pr / 14) % 3, cj = (pr % 14) % 3;
  const int tab[3][3] = {{-4, 1, 2}, {-1, 0, 3}, {-2, -3, 4}};
  const int s = tab[ci][cj];
  const int srct = (t - s + 8) & 7;
  kv[(size_t)(198 + pr) * 128 * 768 + (size_t)c * 768 + d] =
      kv[(size_t)(2 + pr) * 128 * 768 + (size_t)(bb * 8 + srct) * 768 + d];
}

// ---------------------------------------------------------------------------
extern "C" void kernel_launch(void* const* d_in, const int* in_sizes, int n_in,
                              void* d_out, int out_size, void* d_ws, size_t ws_size,
                              hipStream_t stream) {
  const float* x      = (const float*)d_in[0];
  const float* in_w   = (const float*)d_in[1];
  const float* in_b   = (const float*)d_in[2];
  const float* out_w  = (const float*)d_in[3];
  const float* out_b  = (const float*)d_in[4];
  const float* ln1_g  = (const float*)d_in[5];
  const float* ln1_b  = (const float*)d_in[6];
  const float* ln2_g  = (const float*)d_in[7];
  const float* ln2_b  = (const float*)d_in[8];
  const float* fc_w   = (const float*)d_in[9];
  const float* fc_b   = (const float*)d_in[10];
  const float* proj_w = (const float*)d_in[11];
  const float* proj_b = (const float*)d_in[12];
  const float* t1_w = (const float*)d_in[13]; const float* t1_b = (const float*)d_in[14];
  const float* t2_w = (const float*)d_in[15]; const float* t2_b = (const float*)d_in[16];
  const float* s1_w = (const float*)d_in[17]; const float* s1_b = (const float*)d_in[18];
  const float* s2_w = (const float*)d_in[19]; const float* s2_b = (const float*)d_in[20];
  const float* m1_w = (const float*)d_in[21]; const float* m1_b = (const float*)d_in[22];
  const float* m2_w = (const float*)d_in[23]; const float* m2_b = (const float*)d_in[24];
  float* out = (float*)d_out;

  char* base = (char*)d_ws;
  size_t off = 0;
  auto alloc = [&](size_t bytes) -> void* {
    off = (off + 255) & ~(size_t)255;
    void* p = base + off;
    off += bytes;
    return p;
  };
  typedef _Float16 h16;

  h16* in_w_h   = (h16*)alloc((size_t)2304 * 768 * 2);
  h16* out_w_h  = (h16*)alloc((size_t)768 * 768 * 2);
  h16* fc_w_h   = (h16*)alloc((size_t)3072 * 768 * 2);
  h16* proj_w_h = (h16*)alloc((size_t)768 * 3072 * 2);
  h16* t1_h = (h16*)alloc((size_t)192 * 768 * 2);
  h16* t2_h = (h16*)alloc((size_t)768 * 192 * 2);
  h16* s1_h = (h16*)alloc((size_t)192 * 768 * 2);
  h16* s2_h = (h16*)alloc((size_t)768 * 192 * 2);
  h16* m1_h = (h16*)alloc((size_t)192 * 768 * 2);
  h16* m2_h = (h16*)alloc((size_t)768 * 192 * 2);
  h16*   xtn_h    = (h16*)alloc((size_t)128 * 768 * 2);
  h16*   qkvt_h   = (h16*)alloc((size_t)128 * 2304 * 2);
  h16*   ctxt_h   = (h16*)alloc((size_t)128 * 768 * 2);
  h16*   xtattn_h = (h16*)alloc((size_t)128 * 768 * 2);
  h16*   ht_h     = (h16*)alloc((size_t)128 * 192 * 2);
  float* xt2_f    = (float*)alloc((size_t)128 * 768 * 4);
  const size_t TQ = (size_t)198 * 128, TK = (size_t)394 * 128, T2 = (size_t)197 * 128;
  float* xbuf   = (float*)alloc(TQ * 768 * 4);
  h16*   kv_h   = (h16*)alloc(TK * 768 * 2);
  h16*   q_h    = (h16*)alloc(TQ * 768 * 2);
  h16*   kvp_h  = (h16*)alloc(TK * 1536 * 2);
  h16*   ctx_h  = (h16*)alloc(TQ * 768 * 2);
  float* attn_f = (float*)alloc(TQ * 768 * 4);
  h16*   attn_h = (h16*)alloc(TQ * 768 * 2);
  h16*   hs_h   = (h16*)alloc(TQ * 192 * 2);
  // aliases (lifetimes disjoint)
  h16* hid_h = kvp_h;   // 25216x3072 f16 <= 50432x1536 f16
  h16* xn_h  = q_h;     // 25216x768  f16 <= 25344x768  f16
  h16* hm_h  = hs_h;    // 25216x192  f16 <= 25344x192  f16

  auto conv = [&](const float* s, h16* d, long n) {
    conv_f16_k<<<(unsigned)((n + 255) / 256), 256, 0, stream>>>(s, d, n);
  };
  conv(in_w, in_w_h, 2304L * 768);  conv(out_w, out_w_h, 768L * 768);
  conv(fc_w, fc_w_h, 3072L * 768);  conv(proj_w, proj_w_h, 768L * 3072);
  conv(t1_w, t1_h, 192L * 768);     conv(t2_w, t2_h, 768L * 192);
  conv(s1_w, s1_h, 192L * 768);     conv(s2_w, s2_h, 768L * 192);
  conv(m1_w, m1_h, 192L * 768);     conv(m2_w, m2_h, 768L * 192);

  const float ATTN_SCALE = 0.125f;  // 1/sqrt(64)

  // --- temporal token path ----------------------------------------------
  ln_temporal_k<<<16, 256, 0, stream>>>(x, ln1_g, ln1_b, xtn_h);
  gemm128<0><<<dim3(2304 / 128, 1), 256, 0, stream>>>(
      xtn_h, in_w_h, in_b, nullptr, nullptr, nullptr, qkvt_h, 128, 2304, 768, 1.0f);
  flash_attn<<<dim3(16 * NH_, 1), 128, 0, stream>>>(
      qkvt_h, 2304, qkvt_h + 768, 2304, qkvt_h + 1536, 2304, ctxt_h, 768,
      16, 8, 8, ATTN_SCALE);
  gemm128<0><<<dim3(768 / 128, 1), 256, 0, stream>>>(
      ctxt_h, out_w_h, out_b, nullptr, nullptr, nullptr, xtattn_h, 128, 768, 768, 1.0f);
  gemm64<1><<<dim3(192 / 64, 2), 128, 0, stream>>>(
      xtattn_h, t1_h, t1_b, nullptr, ht_h, 128, 192, 768, 1.0f);
  gemm128<0><<<dim3(768 / 128, 1), 256, 0, stream>>>(
      ht_h, t2_h, t2_b, nullptr, nullptr, xt2_f, nullptr, 128, 768, 192, 1.0f);

  // --- assemble 198-token x, LN1 into kv, patch-shift KV -----------------
  build_xbuf_k<<<(unsigned)((198L * 128 * 768 + 255) / 256), 256, 0, stream>>>(
      x, xt2_f, xbuf);
  ln1_k<<<(unsigned)((TQ + 7) / 8), 256, 0, stream>>>(xbuf, ln1_g, ln1_b, kv_h, (int)TQ);
  shift_k<<<(unsigned)((196L * 128 * 768 + 255) / 256), 256, 0, stream>>>(kv_h);

  // --- QKV projections ---------------------------------------------------
  gemm128<0><<<dim3(768 / 128, (unsigned)(TQ / 128)), 256, 0, stream>>>(
      kv_h, in_w_h, in_b, nullptr, nullptr, nullptr, q_h, (int)TQ, 768, 768, 1.0f);
  gemm128<0><<<dim3(1536 / 128, (unsigned)(TK / 128)), 256, 0, stream>>>(
      kv_h, in_w_h + (size_t)768 * 768, in_b + 768, nullptr, nullptr, nullptr,
      kvp_h, (int)TK, 1536, 768, 1.0f);

  // --- main attention (Tq=198, Tk=394) -----------------------------------
  flash_attn<<<dim3(128 * NH_, 4), 128, 0, stream>>>(
      q_h, 768, kvp_h, 1536, kvp_h + 768, 1536, ctx_h, 768,
      128, 198, 394, ATTN_SCALE);
  gemm128<0><<<dim3(768 / 128, (unsigned)(TQ / 128)), 256, 0, stream>>>(
      ctx_h, out_w_h, out_b, nullptr, nullptr, attn_f, attn_h, (int)TQ, 768, 768, 1.0f);

  // --- spatial adapter (skip=True) + residual into xbuf -------------------
  gemm64<1><<<dim3(192 / 64, (unsigned)(TQ / 64)), 128, 0, stream>>>(
      attn_h, s1_h, s1_b, nullptr, hs_h, (int)TQ, 192, 768, 1.0f);
  gemm128<0><<<dim3(768 / 128, (unsigned)(TQ / 128)), 256, 0, stream>>>(
      hs_h, s2_h, s2_b, attn_f, xbuf, xbuf, nullptr, (int)TQ, 768, 192, 1.0f);

  // --- drop temporal token, LN2, residual base in d_out -------------------
  ln2_copy_k<<<(unsigned)((T2 + 7) / 8), 256, 0, stream>>>(
      xbuf, ln2_g, ln2_b, out, xn_h, (int)T2);

  // --- MLP (SiLU-1.702) + residual ---------------------------------------
  gemm128<2><<<dim3(3072 / 128, (unsigned)(T2 / 128)), 256, 0, stream>>>(
      xn_h, fc_w_h, fc_b, nullptr, nullptr, nullptr, hid_h, (int)T2, 3072, 768, 1.0f);
  gemm128<0><<<dim3(768 / 128, (unsigned)(T2 / 128)), 256, 0, stream>>>(
      hid_h, proj_w_h, proj_b, out, nullptr, out, nullptr, (int)T2, 768, 3072, 1.0f);

  // --- MLP adapter, scaled 0.5, accumulate into d_out ---------------------
  gemm64<1><<<dim3(192 / 64, (unsigned)(T2 / 64)), 128, 0, stream>>>(
      xn_h, m1_h, m1_b, nullptr, hm_h, (int)T2, 192, 768, 1.0f);
  gemm128<0><<<dim3(768 / 128, (unsigned)(T2 / 128)), 256, 0, stream>>>(
      hm_h, m2_h, m2_b, out, nullptr, out, nullptr, (int)T2, 768, 192, 0.5f);
}
